// HymbaLayer_9904194584646
// MI455X (gfx1250) — compile-verified, matrix operand order checked by reading, not run on hardware
//
#include <hip/hip_runtime.h>
#include <cstdint>
#include <cstddef>

// ---------------- CDNA5 WMMA types ----------------
typedef __attribute__((ext_vector_type(16))) _Float16 v16h;
typedef __attribute__((ext_vector_type(8)))  _Float16 v8h;
typedef __attribute__((ext_vector_type(8)))  float    v8f;
typedef __attribute__((ext_vector_type(4)))  unsigned int v4u;
typedef __attribute__((ext_vector_type(8)))  int      v8i;
typedef __attribute__((ext_vector_type(4)))  int      v4i;

__device__ __forceinline__ v8f vzero8() {
  v8f v = {0.f,0.f,0.f,0.f,0.f,0.f,0.f,0.f};
  return v;
}
#define WMMA_F16(a, b, c) \
  __builtin_amdgcn_wmma_f32_16x16x32_f16(false, (a), false, (b), (short)0, (c), false, false)
#define CAT16(lo, hi) \
  __builtin_shufflevector((lo), (hi), 0, 1, 2, 3, 4, 5, 6, 7, 8, 9, 10, 11, 12, 13, 14, 15)

// ---------------- problem dims --------------------
constexpr int DMODEL = 1024;
constexpr int NHEADS = 16;
constexpr int HDIM   = 64;
constexpr int DFF    = 4096;
constexpr int DSTATE = 16;
constexpr int DIN    = 2048;   // D_INNER
constexpr int DTRANK = 64;
constexpr int BATCH  = 4;
constexpr int SEQ    = 1024;
constexpr int MROWS  = BATCH * SEQ;   // 4096

// ---------------- math helpers --------------------
__device__ __forceinline__ float sigmoidf_(float x) { return 1.f / (1.f + __expf(-x)); }
__device__ __forceinline__ float siluf_(float x)    { return x / (1.f + __expf(-x)); }
__device__ __forceinline__ float geluf_(float x)    { return 0.5f * x * (1.f + erff(x * 0.70710678118f)); }
__device__ __forceinline__ float softplusf_(float x){ return (x > 20.f) ? x : log1pf(__expf(x)); }

// ---------------- TDM: async 2D tile load (global -> LDS) ----------------
// Loads a 256-row x 32-f16 tile (row stride lda elements) into LDS with
// 16B padding after each 64B row => LDS row pitch 80B (matches sA[.][40]).
// D# layout per CDNA5 ISA (#8.3-8.5).
__device__ __forceinline__ void tdm_stage_A(const _Float16* gsrc, unsigned lds_off, int lda) {
  unsigned long long ga = (unsigned long long)(uintptr_t)gsrc;
  v4u g0;
  g0[0] = 1u;                                   // count=1, user descriptor
  g0[1] = lds_off;                              // LDS byte address
  g0[2] = (unsigned)ga;                         // global_addr[31:0]
  g0[3] = (unsigned)((ga >> 32) & 0x01FFFFFFu)  // global_addr[56:32]
        | (2u << 30);                           // type = 2 ("image")
  v8i g1;
  g1[0] = (int)(0x10000u        // data_size = 1 (2 bytes)
        | 0x100000u             // pad_enable
        | (3u << 22)            // pad_interval: after 64B
        | (3u << 25));          // pad_amount: 4 DWORDs = 16B
  g1[1] = 0;                    // ab_addr=0 | tensor_dim0[15:0] (dim0 = 1<<20 -> lo16=0)
  g1[2] = 0x10;                 // tensor_dim0[31:16]=0x10 | tensor_dim1[15:0]=0 (dim1=1<<20)
  g1[3] = (int)(0x10u | (32u << 16));   // tensor_dim1[31:16]=0x10 | tile_dim0=32
  g1[4] = 256;                  // tile_dim1 = 256 rows | tile_dim2 = 0
  g1[5] = lda;                  // tensor_dim0_stride[31:0] (elements)
  g1[6] = 0;                    // stride hi | tensor_dim1_stride lo
  g1[7] = 0;
  v4i g2; g2[0] = 0; g2[1] = 0; g2[2] = 0; g2[3] = 0;
  v4i g3; g3[0] = 0; g3[1] = 0; g3[2] = 0; g3[3] = 0;
#if defined(__clang_major__) && (__clang_major__ >= 23)
  v8i gx; gx[0]=0; gx[1]=0; gx[2]=0; gx[3]=0; gx[4]=0; gx[5]=0; gx[6]=0; gx[7]=0;
  __builtin_amdgcn_tensor_load_to_lds(g0, g1, g2, g3, gx, 0);
#else
  __builtin_amdgcn_tensor_load_to_lds(g0, g1, g2, g3, 0);
#endif
}

// ---------------- fp32 -> fp16 convert ------------
__global__ __launch_bounds__(256) void cvt_f16_k(const float* __restrict__ src,
                                                 _Float16* __restrict__ dst, int n) {
  int i = blockIdx.x * 256 + threadIdx.x;
  if (i < n) dst[i] = (_Float16)src[i];
}

// ---------------- LayerNorm -----------------------
__global__ __launch_bounds__(256) void ln_k(const float* __restrict__ x,
                                            const float* __restrict__ g,
                                            const float* __restrict__ be,
                                            _Float16* __restrict__ oh, int D) {
  __shared__ float s1[256], s2[256];
  int row = blockIdx.x, tid = threadIdx.x;
  const float* xr = x + (size_t)row * D;
  float s = 0.f, ss = 0.f;
  for (int i = tid; i < D; i += 256) { float v = xr[i]; s += v; ss += v * v; }
  s1[tid] = s; s2[tid] = ss; __syncthreads();
  for (int w = 128; w > 0; w >>= 1) {
    if (tid < w) { s1[tid] += s1[tid + w]; s2[tid] += s2[tid + w]; }
    __syncthreads();
  }
  float mean = s1[0] / D;
  float var  = s2[0] / D - mean * mean;
  float rstd = rsqrtf(var + 1e-5f);
  for (int i = tid; i < D; i += 256) {
    float v = (xr[i] - mean) * rstd * g[i] + be[i];
    oh[(size_t)row * D + i] = (_Float16)v;
  }
}

// ---------------- generic WMMA GEMM ---------------
// C[M,N] = act(A[M,K](f16, lda) * W[K,N](f16, ldw) + bias) + resid
// block: 256 thr = 8 waves; block tile 256x64; wave tile 32x64; K-step 32.
// A tiles DMA'd by the Tensor Data Mover (wave 0, TENSORcnt); W tiles staged
// transposed by all threads so B-fragments are contiguous ds_load_b128 pairs.
// act: 0 none, 1 gelu(exact), 2 softplus

__device__ __forceinline__ void stage_W(
    const _Float16* __restrict__ W, int ldw,
    int n0, int k0, int N, int tid, _Float16 (*sBt)[40]) {
  const int kk = tid >> 3, nb = (tid & 7) * 8;
  const _Float16* wp = W + (size_t)(k0 + kk) * ldw + n0 + nb;
  v8h wv;
  if (n0 + nb + 8 <= N) {
    wv = *(const v8h*)wp;
  } else {
#pragma unroll
    for (int e = 0; e < 8; e++) wv[e] = (n0 + nb + e < N) ? wp[e] : (_Float16)0.f;
  }
#pragma unroll
  for (int e = 0; e < 8; e++) sBt[nb + e][kk] = wv[e];
}

__global__ __launch_bounds__(256) void gemm_f16_wmma(
    const _Float16* __restrict__ A, int lda,
    const _Float16* __restrict__ W, int ldw,
    const float* __restrict__ bias,
    const float* __restrict__ resid,
    float* __restrict__ Cf,
    _Float16* __restrict__ Ch,
    int M, int N, int K, int act) {
  __shared__ _Float16 sA[2][256][40];   // TDM fills 80B-pitch rows (64B data + 16B pad)
  __shared__ _Float16 sBt[2][64][40];   // transposed W tile: [n][k]

  const int tid = threadIdx.x;
  const int wave = tid >> 5, lane = tid & 31;
  const int lr = lane & 15, lh = lane >> 4;
  const int m0 = blockIdx.y * 256, n0 = blockIdx.x * 64;
  const int r0 = wave * 32;

  const unsigned ldsA0 = (unsigned)(uintptr_t)&sA[0][0][0];
  const unsigned ldsA1 = (unsigned)(uintptr_t)&sA[1][0][0];

  v8f acc0[4], acc1[4];
#pragma unroll
  for (int c = 0; c < 4; c++) { acc0[c] = vzero8(); acc1[c] = vzero8(); }

  // prologue: issue TDM for tile 0 (wave 0) + stage W tile 0 (all threads)
  if (tid < 32) tdm_stage_A(A + (size_t)m0 * lda, ldsA0, lda);
  stage_W(W, ldw, n0, 0, N, tid, sBt[0]);
  __builtin_amdgcn_s_wait_tensorcnt(0);
  __syncthreads();

  int buf = 0;
  for (int k0 = 0; k0 < K; k0 += 32) {
    if (k0 + 32 < K) {
      // async: TDM DMA of next A tile overlaps the WMMAs below
      if (tid < 32)
        tdm_stage_A(A + (size_t)m0 * lda + (k0 + 32), buf ? ldsA0 : ldsA1, lda);
      stage_W(W, ldw, n0, k0 + 32, N, tid, sBt[buf ^ 1]);
      if (k0 + 64 < K)
        __builtin_prefetch(W + (size_t)(k0 + 64 + (tid >> 3)) * ldw + n0 + (tid & 7) * 8, 0, 1);
    }

    // A fragments: rows r0..r0+15 and r0+16..r0+31 (each = 2x ds_load_b128)
    v8h alo0 = *(const v8h*)&sA[buf][r0 + lr][lh * 8];
    v8h ahi0 = *(const v8h*)&sA[buf][r0 + lr][lh * 8 + 16];
    v16h a0 = CAT16(alo0, ahi0);
    v8h alo1 = *(const v8h*)&sA[buf][r0 + 16 + lr][lh * 8];
    v8h ahi1 = *(const v8h*)&sA[buf][r0 + 16 + lr][lh * 8 + 16];
    v16h a1 = CAT16(alo1, ahi1);

#pragma unroll
    for (int c = 0; c < 4; c++) {
      v8h blo = *(const v8h*)&sBt[buf][c * 16 + lr][lh * 8];
      v8h bhi = *(const v8h*)&sBt[buf][c * 16 + lr][lh * 8 + 16];
      v16h bf = CAT16(blo, bhi);
      acc0[c] = WMMA_F16(a0, bf, acc0[c]);
      acc1[c] = WMMA_F16(a1, bf, acc1[c]);
    }
    __builtin_amdgcn_s_wait_tensorcnt(0);  // no-op for non-issuing waves
    __syncthreads();
    buf ^= 1;
  }

  // --- epilogue: bias + activation + residual, dual f32/f16 outputs ---
#pragma unroll
  for (int c = 0; c < 4; c++) {
    int n = n0 + c * 16 + lr;
    if (n < N) {
      float bv = bias ? bias[n] : 0.f;
#pragma unroll
      for (int g = 0; g < 2; g++) {
#pragma unroll
        for (int r = 0; r < 8; r++) {
          int m = m0 + r0 + g * 16 + r + 8 * lh;
          float v = (g == 0 ? acc0[c][r] : acc1[c][r]) + bv;
          if (act == 1) v = geluf_(v);
          else if (act == 2) v = softplusf_(v);
          size_t off = (size_t)m * N + n;
          if (resid) v += resid[off];
          if (Cf) Cf[off] = v;
          if (Ch) Ch[off] = (_Float16)v;
        }
      }
    }
  }
}

// ---------------- flash attention (wave-level, causal) ----------------
__global__ __launch_bounds__(256) void flash_attn_k(
    const _Float16* __restrict__ Q, const _Float16* __restrict__ K,
    const _Float16* __restrict__ V, _Float16* __restrict__ O) {
  __shared__ _Float16 sP[8][16][40];  // wave-private P staging: [query row][key]

  const int wave = threadIdx.x >> 5, lane = threadIdx.x & 31;
  const int lr = lane & 15, lh = lane >> 4;
  const int ntq = SEQ / 16;
  int task = blockIdx.x * 8 + wave;
  int qt = task % ntq; int bh = task / ntq;
  int h = bh % NHEADS; int b = bh / NHEADS;
  int m0 = qt * 16;

  v16h qa0, qa1;
  {
    const _Float16* qp = Q + ((size_t)(b * SEQ + m0 + lr)) * DMODEL + h * HDIM;
    v8h l0 = *(const v8h*)(qp + lh * 8);
    v8h h0 = *(const v8h*)(qp + lh * 8 + 16);
    v8h l1 = *(const v8h*)(qp + 32 + lh * 8);
    v8h h1 = *(const v8h*)(qp + 32 + lh * 8 + 16);
    qa0 = CAT16(l0, h0);
    qa1 = CAT16(l1, h1);
  }

  v8f oo[4];
#pragma unroll
  for (int c = 0; c < 4; c++) oo[c] = vzero8();
  float mi[8], si[8];
#pragma unroll
  for (int r = 0; r < 8; r++) { mi[r] = -1e30f; si[r] = 0.f; }

  for (int kb = 0; kb <= m0 + 15; kb += 32) {
    float pv[2][8];
#pragma unroll
    for (int half = 0; half < 2; half++) {
      int t0 = kb + half * 16;
      const _Float16* kp = K + ((size_t)(b * SEQ + t0 + lr)) * DMODEL + h * HDIM;
      v8h l0 = *(const v8h*)(kp + lh * 8);
      v8h h0 = *(const v8h*)(kp + lh * 8 + 16);
      v8h l1 = *(const v8h*)(kp + 32 + lh * 8);
      v8h h1 = *(const v8h*)(kp + 32 + lh * 8 + 16);
      v16h b0 = CAT16(l0, h0);
      v16h b1 = CAT16(l1, h1);
      v8f s = vzero8();
      s = WMMA_F16(qa0, b0, s);
      s = WMMA_F16(qa1, b1, s);
#pragma unroll
      for (int r = 0; r < 8; r++) {
        int row = m0 + r + 8 * lh;
        int t = t0 + lr;
        float sv = s[r] * 0.125f;               // 1/sqrt(64)
        pv[half][r] = (t <= row) ? sv : -1e30f; // causal mask
      }
    }
#pragma unroll
    for (int r = 0; r < 8; r++) {
      float bm = fmaxf(pv[0][r], pv[1][r]);
      bm = fmaxf(bm, __shfl_xor(bm, 1, 32));
      bm = fmaxf(bm, __shfl_xor(bm, 2, 32));
      bm = fmaxf(bm, __shfl_xor(bm, 4, 32));
      bm = fmaxf(bm, __shfl_xor(bm, 8, 32));
      float mnew  = fmaxf(mi[r], bm);
      float scale = __expf(mi[r] - mnew);
      float p0 = __expf(pv[0][r] - mnew);
      float p1 = __expf(pv[1][r] - mnew);
      float ps = p0 + p1;
      ps += __shfl_xor(ps, 1, 32);
      ps += __shfl_xor(ps, 2, 32);
      ps += __shfl_xor(ps, 4, 32);
      ps += __shfl_xor(ps, 8, 32);
      si[r] = si[r] * scale + ps;
      mi[r] = mnew;
      oo[0][r] *= scale; oo[1][r] *= scale; oo[2][r] *= scale; oo[3][r] *= scale;
      sP[wave][r + 8 * lh][lr]      = (_Float16)p0;
      sP[wave][r + 8 * lh][16 + lr] = (_Float16)p1;
    }
    v8h plo = *(const v8h*)&sP[wave][lr][lh * 8];
    v8h phi = *(const v8h*)&sP[wave][lr][lh * 8 + 16];
    v16h pa = CAT16(plo, phi);
#pragma unroll
    for (int c = 0; c < 4; c++) {
      v16h vb;
#pragma unroll
      for (int j = 0; j < 16; j++) {
        int kk = lh * 8 + (j & 7) + ((j >> 3) << 4);
        vb[j] = V[((size_t)(b * SEQ + kb + kk)) * DMODEL + h * HDIM + c * 16 + lr];
      }
      oo[c] = WMMA_F16(pa, vb, oo[c]);
    }
  }

#pragma unroll
  for (int c = 0; c < 4; c++) {
#pragma unroll
    for (int r = 0; r < 8; r++) {
      int row = m0 + r + 8 * lh;
      float val = oo[c][r] / si[r];
      O[((size_t)(b * SEQ + row)) * DMODEL + h * HDIM + c * 16 + lr] = (_Float16)val;
    }
  }
}

// ---------------- causal depthwise conv (K=4) + SiLU ----------------
__global__ __launch_bounds__(256) void conv_silu_k(
    const float* __restrict__ xz, const float* __restrict__ cw,
    const float* __restrict__ cb, float* __restrict__ xc,
    _Float16* __restrict__ xch) {
  size_t idx = (size_t)blockIdx.x * 256 + threadIdx.x;
  size_t total = (size_t)BATCH * SEQ * DIN;
  if (idx >= total) return;
  int d = (int)(idx % DIN);
  size_t bt = idx / DIN;
  int t = (int)(bt % SEQ);
  int b = (int)(bt / SEQ);
  float acc = cb[d];
#pragma unroll
  for (int j = 0; j < 4; j++) {
    int tt = t - 3 + j;
    if (tt >= 0) acc += xz[((size_t)(b * SEQ + tt)) * (2 * DIN) + d] * cw[d * 4 + j];
  }
  float v = siluf_(acc);
  xc[idx]  = v;
  xch[idx] = (_Float16)v;
}

// ---------------- selective SSM scan (sequential over L) ----------------
__global__ __launch_bounds__(256) void ssm_scan_k(
    const float* __restrict__ xc, const float* __restrict__ dtf,
    const float* __restrict__ xdbc, const float* __restrict__ xz,
    const float* __restrict__ A_log, const float* __restrict__ Dp,
    _Float16* __restrict__ yh) {
  const int chunks = DIN / 256;
  int b = blockIdx.x / chunks;
  int d = (blockIdx.x % chunks) * 256 + threadIdx.x;
  float An[DSTATE], hst[DSTATE];
#pragma unroll
  for (int n = 0; n < DSTATE; n++) { An[n] = -__expf(A_log[d * DSTATE + n]); hst[n] = 0.f; }
  float Dv = Dp[d];
  for (int t = 0; t < SEQ; t++) {
    size_t row = (size_t)(b * SEQ + t);
    float xt  = xc[row * DIN + d];
    float dtt = dtf[row * DIN + d];
    const float* dbc = xdbc + row * 96;
    float y = 0.f;
#pragma unroll
    for (int n = 0; n < DSTATE; n++) {
      float e = __expf(dtt * An[n]);
      hst[n] = e * hst[n] + (xt * dtt) * dbc[64 + n];
      y += dbc[80 + n] * hst[n];
    }
    float z  = xz[row * (2 * DIN) + DIN + d];
    float yv = (y + xt * Dv) * siluf_(z);
    yh[row * DIN + d] = (_Float16)yv;
  }
}

// ---------------- branch mix: x + a*attn + (1-a)*ssm ----------------
__global__ __launch_bounds__(256) void mix_k(
    const float* __restrict__ x, const float* __restrict__ attn,
    const float* __restrict__ ssm, const float* __restrict__ mw,
    float* __restrict__ x2, int total, int D) {
  int idx = blockIdx.x * 256 + threadIdx.x;
  if (idx < total) {
    float a = sigmoidf_(mw[idx % D]);
    x2[idx] = x[idx] + a * attn[idx] + (1.f - a) * ssm[idx];
  }
}

// =====================================================================
extern "C" void kernel_launch(void* const* d_in, const int* in_sizes, int n_in,
                              void* d_out, int out_size, void* d_ws, size_t ws_size,
                              hipStream_t stream) {
  const float* x        = (const float*)d_in[0];
  const float* ln1_w    = (const float*)d_in[1];
  const float* ln1_b    = (const float*)d_in[2];
  const float* ln2_w    = (const float*)d_in[3];
  const float* ln2_b    = (const float*)d_in[4];
  const float* q_w      = (const float*)d_in[5];
  const float* q_b      = (const float*)d_in[6];
  const float* k_w      = (const float*)d_in[7];
  const float* k_b      = (const float*)d_in[8];
  const float* v_w      = (const float*)d_in[9];
  const float* v_b      = (const float*)d_in[10];
  const float* o_w      = (const float*)d_in[11];
  const float* o_b      = (const float*)d_in[12];
  const float* in_proj_w= (const float*)d_in[13];
  const float* conv_w   = (const float*)d_in[14];
  const float* conv_b   = (const float*)d_in[15];
  const float* x_proj_w = (const float*)d_in[16];
  const float* dt_w     = (const float*)d_in[17];
  const float* dt_b     = (const float*)d_in[18];
  const float* A_log    = (const float*)d_in[19];
  const float* D_param  = (const float*)d_in[20];
  const float* ssm_out_w= (const float*)d_in[21];
  const float* mix_w    = (const float*)d_in[22];
  const float* ffn_w1   = (const float*)d_in[23];
  const float* ffn_b1   = (const float*)d_in[24];
  const float* ffn_w2   = (const float*)d_in[25];
  const float* ffn_b2   = (const float*)d_in[26];

  char* ws = (char*)d_ws;
  size_t off = 0;
  auto alloc = [&](size_t bytes) -> void* {
    void* p = ws + off;
    off = (off + bytes + 255) & ~(size_t)255;
    return p;
  };

  const size_t MD  = (size_t)MROWS * DMODEL;
  const size_t MDi = (size_t)MROWS * DIN;
  _Float16* xn_h     = (_Float16*)alloc(MD * 2);
  _Float16* q_h      = (_Float16*)alloc(MD * 2);
  _Float16* k_h      = (_Float16*)alloc(MD * 2);
  _Float16* v_h      = (_Float16*)alloc(MD * 2);
  _Float16* attn_h   = (_Float16*)alloc(MD * 2);
  float*    attnpr_f = (float*)   alloc(MD * 4);
  float*    xz_f     = (float*)   alloc((size_t)MROWS * 2 * DIN * 4);
  float*    xc_f     = (float*)   alloc(MDi * 4);
  _Float16* xc_h     = (_Float16*)alloc(MDi * 2);
  float*    xdbc_f   = (float*)   alloc((size_t)MROWS * 96 * 4);
  _Float16* xdbc_h   = (_Float16*)alloc((size_t)MROWS * 96 * 2);
  float*    dt_f     = (float*)   alloc(MDi * 4);
  _Float16* y_h      = (_Float16*)alloc(MDi * 2);
  float*    ssm_f    = (float*)   alloc(MD * 4);
  float*    x2_f     = (float*)   alloc(MD * 4);
  _Float16* h2_h     = (_Float16*)alloc(MD * 2);
  _Float16* ffn1_h   = (_Float16*)alloc((size_t)MROWS * DFF * 2);
  _Float16* qw_h   = (_Float16*)alloc((size_t)DMODEL * DMODEL * 2);
  _Float16* kw_h   = (_Float16*)alloc((size_t)DMODEL * DMODEL * 2);
  _Float16* vw_h   = (_Float16*)alloc((size_t)DMODEL * DMODEL * 2);
  _Float16* ow_h   = (_Float16*)alloc((size_t)DMODEL * DMODEL * 2);
  _Float16* ipw_h  = (_Float16*)alloc((size_t)DMODEL * 2 * DIN * 2);
  _Float16* xpw_h  = (_Float16*)alloc((size_t)DIN * 96 * 2);
  _Float16* dtw_h  = (_Float16*)alloc((size_t)DTRANK * DIN * 2);
  _Float16* sow_h  = (_Float16*)alloc((size_t)DIN * DMODEL * 2);
  _Float16* w1_h   = (_Float16*)alloc((size_t)DMODEL * DFF * 2);
  _Float16* w2_h   = (_Float16*)alloc((size_t)DFF * DMODEL * 2);

  auto cvt = [&](const float* s, _Float16* d, int n) {
    cvt_f16_k<<<(n + 255) / 256, 256, 0, stream>>>(s, d, n);
  };
  cvt(q_w,      qw_h,  DMODEL * DMODEL);
  cvt(k_w,      kw_h,  DMODEL * DMODEL);
  cvt(v_w,      vw_h,  DMODEL * DMODEL);
  cvt(o_w,      ow_h,  DMODEL * DMODEL);
  cvt(in_proj_w,ipw_h, DMODEL * 2 * DIN);
  cvt(x_proj_w, xpw_h, DIN * 96);
  cvt(dt_w,     dtw_h, DTRANK * DIN);
  cvt(ssm_out_w,sow_h, DIN * DMODEL);
  cvt(ffn_w1,   w1_h,  DMODEL * DFF);
  cvt(ffn_w2,   w2_h,  DFF * DMODEL);

  auto gemm = [&](const _Float16* A, int lda, const _Float16* W, int ldw,
                  const float* bias, const float* resid, float* Cf, _Float16* Ch,
                  int M, int N, int K, int act) {
    dim3 grid((N + 63) / 64, M / 256);
    gemm_f16_wmma<<<grid, 256, 0, stream>>>(A, lda, W, ldw, bias, resid, Cf, Ch, M, N, K, act);
  };

  ln_k<<<MROWS, 256, 0, stream>>>(x, ln1_w, ln1_b, xn_h, DMODEL);

  gemm(xn_h, DMODEL, qw_h, DMODEL, q_b, nullptr, nullptr, q_h, MROWS, DMODEL, DMODEL, 0);
  gemm(xn_h, DMODEL, kw_h, DMODEL, k_b, nullptr, nullptr, k_h, MROWS, DMODEL, DMODEL, 0);
  gemm(xn_h, DMODEL, vw_h, DMODEL, v_b, nullptr, nullptr, v_h, MROWS, DMODEL, DMODEL, 0);

  flash_attn_k<<<(BATCH * NHEADS * (SEQ / 16)) / 8, 256, 0, stream>>>(q_h, k_h, v_h, attn_h);

  gemm(attn_h, DMODEL, ow_h, DMODEL, o_b, nullptr, attnpr_f, nullptr, MROWS, DMODEL, DMODEL, 0);
  gemm(xn_h, DMODEL, ipw_h, 2 * DIN, nullptr, nullptr, xz_f, nullptr, MROWS, 2 * DIN, DMODEL, 0);

  conv_silu_k<<<(int)(((size_t)BATCH * SEQ * DIN + 255) / 256), 256, 0, stream>>>(
      xz_f, conv_w, conv_b, xc_f, xc_h);

  gemm(xc_h, DIN, xpw_h, 96, nullptr, nullptr, xdbc_f, xdbc_h, MROWS, 96, DIN, 0);
  gemm(xdbc_h, 96, dtw_h, DIN, dt_b, nullptr, dt_f, nullptr, MROWS, DIN, DTRANK, 2);

  ssm_scan_k<<<BATCH * (DIN / 256), 256, 0, stream>>>(
      xc_f, dt_f, xdbc_f, xz_f, A_log, D_param, y_h);

  gemm(y_h, DIN, sow_h, DMODEL, nullptr, nullptr, ssm_f, nullptr, MROWS, DMODEL, DIN, 0);

  mix_k<<<(int)(MD / 256), 256, 0, stream>>>(x, attnpr_f, ssm_f, mix_w, x2_f, (int)MD, DMODEL);

  ln_k<<<MROWS, 256, 0, stream>>>(x2_f, ln2_w, ln2_b, h2_h, DMODEL);

  gemm(h2_h, DMODEL, w1_h, DFF, ffn_b1, nullptr, nullptr, ffn1_h, MROWS, DFF, DMODEL, 1);
  gemm(ffn1_h, DFF, w2_h, DMODEL, ffn_b2, x2_f, (float*)d_out, nullptr, MROWS, DMODEL, DFF, 0);

  (void)in_sizes; (void)n_in; (void)out_size; (void)ws_size;
}
